// TerraTNTAutoregV6_90108413870868
// MI455X (gfx1250) — compile-verified, matrix operand order checked by reading, not run on hardware
//
#include <hip/hip_runtime.h>
#include <hip/hip_bf16.h>

#define BATCH 256

typedef __attribute__((ext_vector_type(16))) _Float16 v16h;
typedef __attribute__((ext_vector_type(8)))  float    v8f;

// ---------------------------------------------------------------------------
// WMMA helpers (CDNA5: V_WMMA_F32_16X16X32_F16, wave32, D=A(16x32)*B(32x16)+C)
// ---------------------------------------------------------------------------
__device__ inline v8f wmma_f16(v16h a, v16h b, v8f c) {
  return __builtin_amdgcn_wmma_f32_16x16x32_f16(false, a, false, b, (short)0, c,
                                                false, false);
}
__device__ inline v8f vzero8() {
  v8f z;
#pragma unroll
  for (int i = 0; i < 8; ++i) z[i] = 0.f;
  return z;
}

// Optimization barrier on a (wave-uniform) pointer: blocks LICM from hoisting
// the dependent weight loads out of the recurrence loop (which caused massive
// VGPR spill/scratch-reload in the previous build).
__device__ inline const _Float16* opq(const _Float16* p) {
  unsigned long long u = (unsigned long long)(uintptr_t)p;
  asm volatile("" : "+s"(u));
  return (const _Float16*)(uintptr_t)u;
}

// A fragment (16x32 f16) from row-major LDS [16][ld] at column kbase.
// Layout (ISA 7.12.2): lane l: M=l&15, half=l>>4; elem j<8 -> K=kbase+half*8+j,
// elem j>=8 -> K=kbase+half*8+16+(j-8).
__device__ inline v16h frag_a_lds(const _Float16* __restrict__ A, int ld,
                                  int kbase, int lane) {
  int m = lane & 15, half = lane >> 4;
  const _Float16* row = A + m * ld + kbase + half * 8;
  v16h r;
#pragma unroll
  for (int j = 0; j < 8; ++j) { r[j] = row[j]; r[j + 8] = row[j + 16]; }
  return r;
}

// B fragment (32x16 f16) from row-major [N][ldk] f16 weights (K-major rows):
// lane l: N=nbase+(l&15); same K striping as A. Two contiguous 16B runs/lane.
__device__ inline v16h frag_b_glob(const _Float16* __restrict__ W, int ldk,
                                   int nbase, int kbase, int lane) {
  const _Float16* row =
      W + (size_t)(nbase + (lane & 15)) * ldk + kbase + (lane >> 4) * 8;
  v16h r;
#pragma unroll
  for (int j = 0; j < 8; ++j) { r[j] = row[j]; r[j + 8] = row[j + 16]; }
  return r;
}

__device__ inline float fsig(float x) { return 1.f / (1.f + __expf(-x)); }
__device__ inline float ftanh(float x) {
  float e = __expf(-2.f * x);
  return (1.f - e) / (1.f + e);
}

// ---------------------------------------------------------------------------
// Weight packing: f32 -> f16 (optionally padded / column-sliced)
// ---------------------------------------------------------------------------
__global__ void pack_f16(const float* __restrict__ s, _Float16* __restrict__ d,
                         int n) {
  int i = blockIdx.x * blockDim.x + threadIdx.x;
  if (i < n) d[i] = (_Float16)s[i];
}
__global__ void pack_f16_pad(const float* __restrict__ s,
                             _Float16* __restrict__ d, int rows, int cols,
                             int ldp) {
  int i = blockIdx.x * blockDim.x + threadIdx.x;
  if (i < rows * ldp) {
    int r = i / ldp, c = i % ldp;
    d[i] = (_Float16)((c < cols) ? s[r * cols + c] : 0.f);
  }
}
__global__ void pack_slice_f16(const float* __restrict__ s,
                               _Float16* __restrict__ d, int rows, int src_ld,
                               int col0, int cols) {
  int i = blockIdx.x * blockDim.x + threadIdx.x;
  if (i < rows * cols) {
    int r = i / cols, c = i % cols;
    d[i] = (_Float16)s[(size_t)r * src_ld + col0 + c];
  }
}

// ---------------------------------------------------------------------------
// Implicit-GEMM stride-2 3x3 conv + bias + relu. One wave per 16-pixel x
// 16-outchannel tile; K = CIN*9 in 32-wide WMMA chunks (im2col gather).
// SAME padding for stride2/k3: pad_lo=0, pad_hi=1.
// ---------------------------------------------------------------------------
template <int CIN, int COUT, int HIN, int HOUT>
__global__ __launch_bounds__(256) void conv_s2_relu_wmma(
    const float* __restrict__ in, const _Float16* __restrict__ Wh,
    const float* __restrict__ bias, float* __restrict__ out) {
  constexpr int WIN = HIN, WOUT = HOUT;
  constexpr int PIX = HOUT * WOUT;
  constexpr int PT = PIX / 16;
  constexpr int CT = COUT / 16;
  constexpr int K = CIN * 9;
  constexpr int KC = (K + 31) / 32;
  int gw = (blockIdx.x * blockDim.x + threadIdx.x) >> 5;
  int lane = threadIdx.x & 31;
  int tiles = PT * CT;
  int b = gw / tiles;
  if (b >= BATCH) return;  // wave-uniform
  int rem = gw % tiles;
  int pt = rem / CT, ct = rem % CT;
  int m = lane & 15, half = lane >> 4;
  int p = pt * 16 + m;
  int oy = p / WOUT, ox = p % WOUT;
  int co = ct * 16 + m;  // B-fragment N index
  const float* inb = in + (size_t)b * CIN * HIN * WIN;
  v8f acc = vzero8();
  for (int kc = 0; kc < KC; ++kc) {
    v16h a, bf;
#pragma unroll
    for (int j = 0; j < 16; ++j) {
      int k = kc * 32 + (j & 7) + half * 8 + ((j >= 8) ? 16 : 0);
      float av = 0.f, bv = 0.f;
      if (k < K) {
        int ci = k / 9, r9 = k % 9;
        int kh = r9 / 3, kw = r9 % 3;
        int iy = oy * 2 + kh, ix = ox * 2 + kw;
        if (iy < HIN && ix < WIN) av = inb[((size_t)ci * HIN + iy) * WIN + ix];
        bv = (float)Wh[(size_t)co * K + k];
      }
      a[j] = (_Float16)av;
      bf[j] = (_Float16)bv;
    }
    acc = wmma_f16(a, bf, acc);
  }
  float* outb = out + (size_t)b * COUT * PIX;
  int n = ct * 16 + (lane & 15);
  float bs = bias[n];
#pragma unroll
  for (int r = 0; r < 8; ++r) {
    int mm = r + half * 8;
    int pp = pt * 16 + mm;
    outb[(size_t)n * PIX + pp] = fmaxf(acc[r] + bs, 0.f);
  }
}

__global__ __launch_bounds__(128) void global_mean8x8(
    const float* __restrict__ es, float* __restrict__ eg) {
  int b = blockIdx.x, c = threadIdx.x;  // 128 threads
  const float* p = es + ((size_t)b * 128 + c) * 64;
  float s = 0.f;
#pragma unroll
  for (int i = 0; i < 64; ++i) s += p[i];
  eg[b * 128 + c] = s * (1.f / 64.f);
}

// ---------------------------------------------------------------------------
// History 2-layer LSTM, H=128, 60 steps. One block per 16-batch tile.
// Gates (16x512) via WMMA per layer; state in dynamic LDS.
// ---------------------------------------------------------------------------
__global__ __launch_bounds__(256, 1) void hist_lstm(
    const float* __restrict__ hist,                      // B,60,26
    const _Float16* __restrict__ Wih0,                   // 512x32
    const _Float16* __restrict__ Whh0,                   // 512x128
    const _Float16* __restrict__ Wih1,                   // 512x128
    const _Float16* __restrict__ Whh1,                   // 512x128
    const float* __restrict__ bih0, const float* __restrict__ bhh0,
    const float* __restrict__ bih1, const float* __restrict__ bhh1,
    float* __restrict__ hf) {                            // B x 128
  extern __shared__ char smem[];
  _Float16* xh = (_Float16*)smem;        // 16x32
  _Float16* h0h = xh + 16 * 32;          // 16x128
  _Float16* h1h = h0h + 16 * 128;        // 16x128
  float* c0 = (float*)(h1h + 16 * 128);  // 16x128
  float* c1 = c0 + 16 * 128;             // 16x128
  float* g = c1 + 16 * 128;              // 16x512
  float* bs0 = g + 16 * 512;             // 512 (bih0+bhh0)
  float* bs1 = bs0 + 512;                // 512 (bih1+bhh1)
  int tid = threadIdx.x, lane = tid & 31, wave = tid >> 5;
  int b0 = blockIdx.x * 16;
  for (int i = tid; i < 16 * 128; i += 256) {
    c0[i] = 0.f; c1[i] = 0.f;
    h0h[i] = (_Float16)0.f; h1h[i] = (_Float16)0.f;
  }
  for (int i = tid; i < 512; i += 256) {
    bs0[i] = bih0[i] + bhh0[i];
    bs1[i] = bih1[i] + bhh1[i];
  }
  __syncthreads();
  for (int t = 0; t < 60; ++t) {
    for (int i = tid; i < 16 * 32; i += 256) {
      int mb = i >> 5, c = i & 31;
      xh[i] = (_Float16)((c < 26) ? hist[((size_t)(b0 + mb) * 60 + t) * 26 + c]
                                  : 0.f);
    }
    __syncthreads();
    // layer0: g = x@Wih0^T + h0@Whh0^T  (N=512 -> 32 tiles, 4 per wave)
    for (int nt = wave * 4; nt < wave * 4 + 4; ++nt) {
      const _Float16* Wx = opq(Wih0);
      const _Float16* Wr = opq(Whh0);
      v8f acc = vzero8();
      acc = wmma_f16(frag_a_lds(xh, 32, 0, lane),
                     frag_b_glob(Wx, 32, nt * 16, 0, lane), acc);
      for (int kc = 0; kc < 4; ++kc)
        acc = wmma_f16(frag_a_lds(h0h, 128, kc * 32, lane),
                       frag_b_glob(Wr, 128, nt * 16, kc * 32, lane), acc);
      int n = nt * 16 + (lane & 15);
#pragma unroll
      for (int r = 0; r < 8; ++r) g[(r + (lane >> 4) * 8) * 512 + n] = acc[r];
    }
    __syncthreads();
    for (int i = tid; i < 16 * 128; i += 256) {
      int mb = i >> 7, j = i & 127;
      const float* gr = g + mb * 512;
      float ig = gr[j] + bs0[j];
      float fg = gr[128 + j] + bs0[128 + j];
      float gg = gr[256 + j] + bs0[256 + j];
      float og = gr[384 + j] + bs0[384 + j];
      float c = fsig(fg) * c0[i] + fsig(ig) * ftanh(gg);
      c0[i] = c;
      h0h[i] = (_Float16)(fsig(og) * ftanh(c));
    }
    __syncthreads();
    // layer1
    for (int nt = wave * 4; nt < wave * 4 + 4; ++nt) {
      const _Float16* Wx = opq(Wih1);
      const _Float16* Wr = opq(Whh1);
      v8f acc = vzero8();
      for (int kc = 0; kc < 4; ++kc)
        acc = wmma_f16(frag_a_lds(h0h, 128, kc * 32, lane),
                       frag_b_glob(Wx, 128, nt * 16, kc * 32, lane), acc);
      for (int kc = 0; kc < 4; ++kc)
        acc = wmma_f16(frag_a_lds(h1h, 128, kc * 32, lane),
                       frag_b_glob(Wr, 128, nt * 16, kc * 32, lane), acc);
      int n = nt * 16 + (lane & 15);
#pragma unroll
      for (int r = 0; r < 8; ++r) g[(r + (lane >> 4) * 8) * 512 + n] = acc[r];
    }
    __syncthreads();
    for (int i = tid; i < 16 * 128; i += 256) {
      int mb = i >> 7, j = i & 127;
      const float* gr = g + mb * 512;
      float ig = gr[j] + bs1[j];
      float fg = gr[128 + j] + bs1[128 + j];
      float gg = gr[256 + j] + bs1[256 + j];
      float og = gr[384 + j] + bs1[384 + j];
      float c = fsig(fg) * c1[i] + fsig(ig) * ftanh(gg);
      c1[i] = c;
      float h = fsig(og) * ftanh(c);
      h1h[i] = (_Float16)h;
      if (t == 59) hf[(size_t)(b0 + mb) * 128 + j] = h;
    }
    __syncthreads();
  }
}

// ---------------------------------------------------------------------------
// Goal scoring (VALU) + argmax + goal_fc + fusion (WMMA) -> context
// ---------------------------------------------------------------------------
__global__ __launch_bounds__(256) void goal_context(
    const float* __restrict__ eg, const float* __restrict__ hf,
    const float* __restrict__ cand,                       // B,6,2
    const float* __restrict__ gc_w1, const float* __restrict__ gc_b1,
    const float* __restrict__ gc_w2, const float* __restrict__ gc_b2,
    const float* __restrict__ gfw, const float* __restrict__ gfb,  // 64x2, 64
    const _Float16* __restrict__ fus_h,                   // 256x320 f16
    const float* __restrict__ fus_b,
    float* __restrict__ logits_out,                       // B x 6
    float* __restrict__ goal_sel,                         // B x 2
    float* __restrict__ ctx_f32, _Float16* __restrict__ ctx_f16) {
  __shared__ float ctxg[16][256];    // [eg(0:128) | hf(128:256)] per batch row
  __shared__ float logitsL[16][6];
  __shared__ float selg[16][2];
  __shared__ float gfeat[16][64];
  __shared__ _Float16 fA[16][320];   // [hf | eg | gfeat]
  int tid = threadIdx.x, lane = tid & 31, wave = tid >> 5;
  int b0 = blockIdx.x * 16;
  for (int i = tid; i < 16 * 256; i += 256) {
    int mb = i >> 8, c = i & 255;
    ctxg[mb][c] = (c < 128) ? eg[(b0 + mb) * 128 + c]
                            : hf[(b0 + mb) * 128 + (c - 128)];
  }
  __syncthreads();
  // goal scores: 96 (b,cand) pairs, 12 per wave; lane covers 4 hidden dims
  for (int q = 0; q < 12; ++q) {
    int pi = wave * 12 + q;
    int mb = pi / 6, cd = pi % 6;
    float cx = cand[((b0 + mb) * 6 + cd) * 2 + 0] * (1.f / 70.f);
    float cy = cand[((b0 + mb) * 6 + cd) * 2 + 1] * (1.f / 70.f);
    float part = 0.f;
#pragma unroll
    for (int dd = 0; dd < 4; ++dd) {
      int d = lane + dd * 32;
      const float* w = gc_w1 + (size_t)d * 258;
      float s = gc_b1[d];
      for (int k = 0; k < 256; ++k) s += w[k] * ctxg[mb][k];
      s += w[256] * cx + w[257] * cy;
      part += gc_w2[d] * fmaxf(s, 0.f);
    }
    for (int o = 16; o > 0; o >>= 1) part += __shfl_down(part, o, 32);
    if (lane == 0) {
      float lg = part + gc_b2[0];
      logitsL[mb][cd] = lg;
      logits_out[(b0 + mb) * 6 + cd] = lg;
    }
  }
  __syncthreads();
  if (tid < 16) {
    int best = 0;
    float bv = logitsL[tid][0];
    for (int c = 1; c < 6; ++c)
      if (logitsL[tid][c] > bv) { bv = logitsL[tid][c]; best = c; }
    float gx = cand[((b0 + tid) * 6 + best) * 2 + 0];
    float gy = cand[((b0 + tid) * 6 + best) * 2 + 1];
    selg[tid][0] = gx; selg[tid][1] = gy;
    goal_sel[(b0 + tid) * 2 + 0] = gx;
    goal_sel[(b0 + tid) * 2 + 1] = gy;
  }
  __syncthreads();
  for (int i = tid; i < 16 * 64; i += 256) {
    int mb = i >> 6, d = i & 63;
    gfeat[mb][d] =
        fmaxf(gfw[d * 2] * selg[mb][0] + gfw[d * 2 + 1] * selg[mb][1] + gfb[d],
              0.f);
  }
  __syncthreads();
  for (int i = tid; i < 16 * 320; i += 256) {
    int mb = i / 320, k = i % 320;
    float v = (k < 128) ? ctxg[mb][128 + k]                 // history
              : (k < 256) ? ctxg[mb][k - 128]               // env_global
                          : gfeat[mb][k - 256];             // goal feature
    fA[mb][k] = (_Float16)v;
  }
  __syncthreads();
  // fusion: [16x320]@[320x256] -> relu -> context  (16 N-tiles, 2 per wave)
  for (int nt = wave * 2; nt < wave * 2 + 2; ++nt) {
    v8f acc = vzero8();
    for (int kc = 0; kc < 10; ++kc)
      acc = wmma_f16(frag_a_lds(&fA[0][0], 320, kc * 32, lane),
                     frag_b_glob(fus_h, 320, nt * 16, kc * 32, lane), acc);
    int n = nt * 16 + (lane & 15);
    float bs = fus_b[n];
#pragma unroll
    for (int r = 0; r < 8; ++r) {
      int mb = r + (lane >> 4) * 8;
      float v = fmaxf(acc[r] + bs, 0.f);
      ctx_f32[(size_t)(b0 + mb) * 256 + n] = v;
      ctx_f16[(size_t)(b0 + mb) * 256 + n] = (_Float16)v;
    }
  }
}

// ---------------------------------------------------------------------------
// Waypoints + bilinear sampling of env_spatial + segment conditioners (VALU;
// tiny FLOPs). One block per batch element.
// ---------------------------------------------------------------------------
__global__ __launch_bounds__(256) void waypoints_seg(
    const float* __restrict__ ctx, const float* __restrict__ goal_sel,
    const float* __restrict__ wpq,                        // 10x256
    const float* __restrict__ wtw, const float* __restrict__ wtb,  // 256
    const float* __restrict__ wow, const float* __restrict__ wob,  // 2x256, 2
    const float* __restrict__ res_scale, const float* __restrict__ es,
    const float* __restrict__ spw, const float* __restrict__ spb,  // 256x128
    const float* __restrict__ env_scale,
    const float* __restrict__ segw, const float* __restrict__ segb, // 256x260
    _Float16* __restrict__ seg_h) {                       // B,10,256 f16
  __shared__ float nodes[11][2];
  __shared__ float samp[11][128];
  __shared__ float wpe[11][256];
  __shared__ float red0[256], red1[256];
  int b = blockIdx.x, tid = threadIdx.x;
  if (tid < 2) nodes[0][tid] = 0.f;
  float gx = goal_sel[b * 2 + 0], gy = goal_sel[b * 2 + 1];
  float scale = res_scale[0];
  for (int i = 0; i < 10; ++i) {
    int wpi = (i < 9) ? (32 * (i + 1) - 1) : 359;
    float frac = (float)(wpi + 1) / 360.f;
    float tf = fmaxf(frac * wtw[tid] + wtb[tid], 0.f);
    float wh = ctx[(size_t)b * 256 + tid] + wpq[i * 256 + tid] + tf;
    red0[tid] = wh * wow[tid];
    red1[tid] = wh * wow[256 + tid];
    __syncthreads();
    for (int s = 128; s > 0; s >>= 1) {
      if (tid < s) { red0[tid] += red0[tid + s]; red1[tid] += red1[tid + s]; }
      __syncthreads();
    }
    if (tid == 0) {
      nodes[i + 1][0] = gx * frac + ftanh(red0[0] + wob[0]) * scale;
      nodes[i + 1][1] = gy * frac + ftanh(red1[0] + wob[1]) * scale;
    }
    __syncthreads();
  }
  // bilinear sample 8x8 env_spatial at 11 nodes
  const float* esb = es + (size_t)b * 128 * 64;
  for (int i = tid; i < 11 * 128; i += 256) {
    int nd = i / 128, c = i % 128;
    float ngx = fminf(fmaxf(nodes[nd][0] * (1.f / 70.f), -1.f), 1.f);
    float ngy = fminf(fmaxf(-nodes[nd][1] * (1.f / 70.f), -1.f), 1.f);
    float ix = (ngx + 1.f) * 0.5f * 7.f;
    float iy = (ngy + 1.f) * 0.5f * 7.f;
    int x0 = (int)floorf(ix), y0 = (int)floorf(iy);
    float wx = ix - (float)x0, wy = iy - (float)y0;
    int x0c = min(max(x0, 0), 7), y0c = min(max(y0, 0), 7);
    int x1 = min(max(x0 + 1, 0), 7), y1 = min(max(y0 + 1, 0), 7);
    const float* ch = esb + c * 64;
    float f00 = ch[y0c * 8 + x0c], f01 = ch[y0c * 8 + x1];
    float f10 = ch[y1 * 8 + x0c], f11 = ch[y1 * 8 + x1];
    samp[nd][c] = f00 * (1.f - wx) * (1.f - wy) + f01 * wx * (1.f - wy) +
                  f10 * (1.f - wx) * wy + f11 * wx * wy;
  }
  __syncthreads();
  for (int i = tid; i < 11 * 256; i += 256) {
    int nd = i / 256, d = i % 256;
    const float* w = spw + (size_t)d * 128;
    float s = spb[d];
    for (int k = 0; k < 128; ++k) s += w[k] * samp[nd][k];
    wpe[nd][d] = fmaxf(s, 0.f);
  }
  __syncthreads();
  float escale = env_scale[0];
  for (int sseg = 0; sseg < 10; ++sseg) {
    int d = tid;
    const float* w = segw + (size_t)d * 260;
    float s = segb[d];
    s += w[0] * (nodes[sseg][0] * 0.02f) + w[1] * (nodes[sseg][1] * 0.02f);
    s += w[2] * (nodes[sseg + 1][0] * 0.02f) + w[3] * (nodes[sseg + 1][1] * 0.02f);
    for (int k = 0; k < 256; ++k)
      s += w[4 + k] * 0.5f * escale * (wpe[sseg][k] + wpe[sseg + 1][k]);
    seg_h[((size_t)b * 10 + sseg) * 256 + d] = (_Float16)fmaxf(s, 0.f);
  }
}

// ---------------------------------------------------------------------------
// Decoder: 2-layer LSTM, H=256, 360 steps. One block per 16-batch tile,
// 8 waves. Per step: 512 (layer0) + 1024 (layer1) WMMAs per block. The
// context+segment+bias contribution to layer0 gates is precomputed per
// segment (changes 10x over 360 steps) into LDS.
// ---------------------------------------------------------------------------
__global__ __launch_bounds__(256, 1) void decoder_lstm(
    const _Float16* __restrict__ ctxh_g,                  // B x 256
    const _Float16* __restrict__ segh,                    // B,10,256
    const _Float16* __restrict__ W0pos,                   // 1024x2
    const _Float16* __restrict__ W0ctx,                   // 1024x256
    const _Float16* __restrict__ W0st,                    // 1024x256
    const _Float16* __restrict__ Whh0,                    // 1024x256
    const _Float16* __restrict__ Wih1,                    // 1024x256
    const _Float16* __restrict__ Whh1,                    // 1024x256
    const float* __restrict__ bih0, const float* __restrict__ bhh0,
    const float* __restrict__ bih1, const float* __restrict__ bhh1,
    const float* __restrict__ ow, const float* __restrict__ ob,  // 2x256, 2
    float* __restrict__ pred) {                           // B,360,2
  extern __shared__ char smem[];
  float* sbase = (float*)smem;            // 16x1024
  float* g = sbase + 16 * 1024;           // 16x1024
  float* c0 = g + 16 * 1024;              // 16x256
  float* c1 = c0 + 16 * 256;              // 16x256
  float* pos = c1 + 16 * 256;             // 16x2
  float* bsum0 = pos + 32;                // 1024 (bih0+bhh0)
  float* bsum1 = bsum0 + 1024;            // 1024 (bih1+bhh1)
  float* red = bsum1 + 1024;              // 256 (pos-update partials)
  _Float16* h0h = (_Float16*)(red + 256); // 16x256
  _Float16* h1h = h0h + 16 * 256;
  _Float16* cth = h1h + 16 * 256;
  _Float16* sth = cth + 16 * 256;
  int tid = threadIdx.x, lane = tid & 31, wave = tid >> 5;
  int b0 = blockIdx.x * 16;
  for (int i = tid; i < 16 * 256; i += 256) {
    c0[i] = 0.f; c1[i] = 0.f;
    h0h[i] = (_Float16)0.f;
    _Float16 cv = ctxh_g[(size_t)(b0 + (i >> 8)) * 256 + (i & 255)];
    cth[i] = cv;
    h1h[i] = cv;  // h1 initialized to context (reference init carry)
  }
  for (int i = tid; i < 1024; i += 256) {
    bsum0[i] = bih0[i] + bhh0[i];
    bsum1[i] = bih1[i] + bhh1[i];
  }
  if (tid < 32) pos[tid] = 0.f;
  __syncthreads();
  int segp = -1;
  for (int t = 0; t < 360; ++t) {
    int seg = t >> 5; if (seg > 9) seg = 9;
    if (seg != segp) {
      segp = seg;
      for (int i = tid; i < 16 * 256; i += 256)
        sth[i] = segh[((size_t)(b0 + (i >> 8)) * 10 + seg) * 256 + (i & 255)];
      __syncthreads();
      // sbase = ctx@W0ctx^T + st@W0st^T + bih0 + bhh0
      for (int nt = wave * 8; nt < wave * 8 + 8; ++nt) {
        const _Float16* Wc = opq(W0ctx);
        const _Float16* Ws = opq(W0st);
        v8f acc = vzero8();
        for (int kc = 0; kc < 8; ++kc)
          acc = wmma_f16(frag_a_lds(cth, 256, kc * 32, lane),
                         frag_b_glob(Wc, 256, nt * 16, kc * 32, lane), acc);
        for (int kc = 0; kc < 8; ++kc)
          acc = wmma_f16(frag_a_lds(sth, 256, kc * 32, lane),
                         frag_b_glob(Ws, 256, nt * 16, kc * 32, lane), acc);
        int n = nt * 16 + (lane & 15);
        float bs = bsum0[n];
#pragma unroll
        for (int r = 0; r < 8; ++r)
          sbase[(r + (lane >> 4) * 8) * 1024 + n] = acc[r] + bs;
      }
      __syncthreads();
    }
    // layer0: g = sbase + pos@W0pos^T + h0@Whh0^T
    for (int nt = wave * 8; nt < wave * 8 + 8; ++nt) {
      const _Float16* Wr = opq(Whh0);
      const _Float16* Wp = opq(W0pos);
      v8f acc = vzero8();
      for (int kc = 0; kc < 8; ++kc)
        acc = wmma_f16(frag_a_lds(h0h, 256, kc * 32, lane),
                       frag_b_glob(Wr, 256, nt * 16, kc * 32, lane), acc);
      int n = nt * 16 + (lane & 15);
      float w0 = (float)Wp[n * 2], w1 = (float)Wp[n * 2 + 1];
#pragma unroll
      for (int r = 0; r < 8; ++r) {
        int mb = r + (lane >> 4) * 8;
        g[mb * 1024 + n] =
            sbase[mb * 1024 + n] + acc[r] + pos[mb * 2] * w0 + pos[mb * 2 + 1] * w1;
      }
    }
    __syncthreads();
    for (int i = tid; i < 16 * 256; i += 256) {
      int mb = i >> 8, j = i & 255;
      const float* gr = g + mb * 1024;
      float c = fsig(gr[256 + j]) * c0[i] + fsig(gr[j]) * ftanh(gr[512 + j]);
      c0[i] = c;
      h0h[i] = (_Float16)(fsig(gr[768 + j]) * ftanh(c));
    }
    __syncthreads();
    // layer1: g = h0@Wih1^T + h1@Whh1^T + biases
    for (int nt = wave * 8; nt < wave * 8 + 8; ++nt) {
      const _Float16* Wx = opq(Wih1);
      const _Float16* Wr = opq(Whh1);
      v8f acc = vzero8();
      for (int kc = 0; kc < 8; ++kc)
        acc = wmma_f16(frag_a_lds(h0h, 256, kc * 32, lane),
                       frag_b_glob(Wx, 256, nt * 16, kc * 32, lane), acc);
      for (int kc = 0; kc < 8; ++kc)
        acc = wmma_f16(frag_a_lds(h1h, 256, kc * 32, lane),
                       frag_b_glob(Wr, 256, nt * 16, kc * 32, lane), acc);
      int n = nt * 16 + (lane & 15);
      float bs = bsum1[n];
#pragma unroll
      for (int r = 0; r < 8; ++r)
        g[(r + (lane >> 4) * 8) * 1024 + n] = acc[r] + bs;
    }
    __syncthreads();
    for (int i = tid; i < 16 * 256; i += 256) {
      int mb = i >> 8, j = i & 255;
      const float* gr = g + mb * 1024;
      float c = fsig(gr[256 + j]) * c1[i] + fsig(gr[j]) * ftanh(gr[512 + j]);
      c1[i] = c;
      h1h[i] = (_Float16)(fsig(gr[768 + j]) * ftanh(c));
    }
    __syncthreads();
    // pos update: 256 threads compute 32-wide partials, 32 threads finish.
    {
      int mb = tid >> 4, d = (tid >> 3) & 1, ch = tid & 7;
      const float* wr = ow + d * 256;
      float s = 0.f;
      for (int k = ch * 32; k < ch * 32 + 32; ++k)
        s += (float)h1h[mb * 256 + k] * wr[k];
      red[tid] = s;
    }
    __syncthreads();
    if (tid < 32) {
      int mb = tid >> 1, d = tid & 1;
      float tot = ob[d];
#pragma unroll
      for (int c = 0; c < 8; ++c) tot += red[(mb << 4) + (d << 3) + c];
      float np = pos[mb * 2 + d] + tot;
      pos[mb * 2 + d] = np;
      pred[((size_t)(b0 + mb) * 360 + t) * 2 + d] = np;
    }
    __syncthreads();
  }
}

// ---------------------------------------------------------------------------
// Host launcher
// ---------------------------------------------------------------------------
static inline int cdiv_i(size_t n, int b) { return (int)((n + b - 1) / b); }

extern "C" void kernel_launch(void* const* d_in, const int* in_sizes, int n_in,
                              void* d_out, int out_size, void* d_ws,
                              size_t ws_size, hipStream_t stream) {
  (void)in_sizes; (void)n_in; (void)out_size; (void)ws_size;
  const float* env_map = (const float*)d_in[0];
  const float* history = (const float*)d_in[1];
  const float* cand = (const float*)d_in[2];
  const float* conv1_w = (const float*)d_in[4];
  const float* conv1_b = (const float*)d_in[5];
  const float* conv2_w = (const float*)d_in[6];
  const float* conv2_b = (const float*)d_in[7];
  const float* conv3_w = (const float*)d_in[8];
  const float* conv3_b = (const float*)d_in[9];
  const float* hWih0 = (const float*)d_in[10];
  const float* hWhh0 = (const float*)d_in[11];
  const float* hbih0 = (const float*)d_in[12];
  const float* hbhh0 = (const float*)d_in[13];
  const float* hWih1 = (const float*)d_in[14];
  const float* hWhh1 = (const float*)d_in[15];
  const float* hbih1 = (const float*)d_in[16];
  const float* hbhh1 = (const float*)d_in[17];
  const float* gc_w1 = (const float*)d_in[18];
  const float* gc_b1 = (const float*)d_in[19];
  const float* gc_w2 = (const float*)d_in[20];
  const float* gc_b2 = (const float*)d_in[21];
  const float* gfw = (const float*)d_in[22];
  const float* gfb = (const float*)d_in[23];
  const float* fus_w = (const float*)d_in[24];
  const float* fus_b = (const float*)d_in[25];
  const float* wpq = (const float*)d_in[26];
  const float* wtw = (const float*)d_in[27];
  const float* wtb = (const float*)d_in[28];
  const float* wow = (const float*)d_in[29];
  const float* wob = (const float*)d_in[30];
  const float* res_scale = (const float*)d_in[31];
  const float* spw = (const float*)d_in[32];
  const float* spb = (const float*)d_in[33];
  const float* env_scale = (const float*)d_in[34];
  const float* dWih0 = (const float*)d_in[35];
  const float* dWhh0 = (const float*)d_in[36];
  const float* dbih0 = (const float*)d_in[37];
  const float* dbhh0 = (const float*)d_in[38];
  const float* dWih1 = (const float*)d_in[39];
  const float* dWhh1 = (const float*)d_in[40];
  const float* dbih1 = (const float*)d_in[41];
  const float* dbhh1 = (const float*)d_in[42];
  const float* ow = (const float*)d_in[43];
  const float* ob = (const float*)d_in[44];
  const float* segw = (const float*)d_in[45];
  const float* segb = (const float*)d_in[46];

  // workspace carve (bytes, 256-aligned)
  char* base = (char*)d_ws;
  size_t off = 0;
  auto carve = [&](size_t bytes) {
    char* p = base + off;
    off = (off + bytes + 255) & ~(size_t)255;
    return p;
  };
  float* a1 = (float*)carve((size_t)BATCH * 64 * 32 * 32 * 4);
  float* a2 = (float*)carve((size_t)BATCH * 128 * 16 * 16 * 4);
  float* es = (float*)carve((size_t)BATCH * 128 * 8 * 8 * 4);
  float* eg = (float*)carve((size_t)BATCH * 128 * 4);
  float* hfb = (float*)carve((size_t)BATCH * 128 * 4);
  float* ctxf = (float*)carve((size_t)BATCH * 256 * 4);
  _Float16* ctxh = (_Float16*)carve((size_t)BATCH * 256 * 2);
  float* gsel = (float*)carve((size_t)BATCH * 2 * 4);
  _Float16* seghb = (_Float16*)carve((size_t)BATCH * 10 * 256 * 2);
  _Float16* wc1h = (_Float16*)carve((size_t)64 * 162 * 2);
  _Float16* wc2h = (_Float16*)carve((size_t)128 * 576 * 2);
  _Float16* wc3h = (_Float16*)carve((size_t)128 * 1152 * 2);
  _Float16* wih0h = (_Float16*)carve((size_t)512 * 32 * 2);
  _Float16* whh0h = (_Float16*)carve((size_t)512 * 128 * 2);
  _Float16* wih1h = (_Float16*)carve((size_t)512 * 128 * 2);
  _Float16* whh1h = (_Float16*)carve((size_t)512 * 128 * 2);
  _Float16* fush = (_Float16*)carve((size_t)256 * 320 * 2);
  _Float16* w0pos = (_Float16*)carve((size_t)1024 * 2 * 2);
  _Float16* w0ctx = (_Float16*)carve((size_t)1024 * 256 * 2);
  _Float16* w0st = (_Float16*)carve((size_t)1024 * 256 * 2);
  _Float16* dwhh0 = (_Float16*)carve((size_t)1024 * 256 * 2);
  _Float16* dwih1 = (_Float16*)carve((size_t)1024 * 256 * 2);
  _Float16* dwhh1 = (_Float16*)carve((size_t)1024 * 256 * 2);

  float* pred = (float*)d_out;                          // B*360*2
  float* logits = (float*)d_out + (size_t)BATCH * 720;  // B*6

  // --- pack weights to f16 ---
  pack_f16<<<cdiv_i(64 * 162, 256), 256, 0, stream>>>(conv1_w, wc1h, 64 * 162);
  pack_f16<<<cdiv_i(128 * 576, 256), 256, 0, stream>>>(conv2_w, wc2h, 128 * 576);
  pack_f16<<<cdiv_i(128 * 1152, 256), 256, 0, stream>>>(conv3_w, wc3h, 128 * 1152);
  pack_f16_pad<<<cdiv_i(512 * 32, 256), 256, 0, stream>>>(hWih0, wih0h, 512, 26, 32);
  pack_f16<<<cdiv_i(512 * 128, 256), 256, 0, stream>>>(hWhh0, whh0h, 512 * 128);
  pack_f16<<<cdiv_i(512 * 128, 256), 256, 0, stream>>>(hWih1, wih1h, 512 * 128);
  pack_f16<<<cdiv_i(512 * 128, 256), 256, 0, stream>>>(hWhh1, whh1h, 512 * 128);
  pack_f16<<<cdiv_i(256 * 320, 256), 256, 0, stream>>>(fus_w, fush, 256 * 320);
  pack_slice_f16<<<cdiv_i(1024 * 2, 256), 256, 0, stream>>>(dWih0, w0pos, 1024, 514, 0, 2);
  pack_slice_f16<<<cdiv_i(1024 * 256, 256), 256, 0, stream>>>(dWih0, w0ctx, 1024, 514, 2, 256);
  pack_slice_f16<<<cdiv_i(1024 * 256, 256), 256, 0, stream>>>(dWih0, w0st, 1024, 514, 258, 256);
  pack_f16<<<cdiv_i(1024 * 256, 256), 256, 0, stream>>>(dWhh0, dwhh0, 1024 * 256);
  pack_f16<<<cdiv_i(1024 * 256, 256), 256, 0, stream>>>(dWih1, dwih1, 1024 * 256);
  pack_f16<<<cdiv_i(1024 * 256, 256), 256, 0, stream>>>(dWhh1, dwhh1, 1024 * 256);

  // --- conv encoder (WMMA implicit GEMM) ---
  conv_s2_relu_wmma<18, 64, 64, 32><<<8192, 256, 0, stream>>>(env_map, wc1h, conv1_b, a1);
  conv_s2_relu_wmma<64, 128, 32, 16><<<4096, 256, 0, stream>>>(a1, wc2h, conv2_b, a2);
  conv_s2_relu_wmma<128, 128, 16, 8><<<1024, 256, 0, stream>>>(a2, wc3h, conv3_b, es);
  global_mean8x8<<<BATCH, 128, 0, stream>>>(es, eg);

  // --- history LSTM ---
  size_t hist_smem = (16 * 32 + 2 * 16 * 128) * sizeof(_Float16) +
                     (2 * 16 * 128 + 16 * 512 + 1024) * sizeof(float);
  hist_lstm<<<BATCH / 16, 256, hist_smem, stream>>>(
      history, wih0h, whh0h, wih1h, whh1h, hbih0, hbhh0, hbih1, hbhh1, hfb);

  // --- goal head + fusion context ---
  goal_context<<<BATCH / 16, 256, 0, stream>>>(
      eg, hfb, cand, gc_w1, gc_b1, gc_w2, gc_b2, gfw, gfb, fush, fus_b, logits,
      gsel, ctxf, ctxh);

  // --- waypoints + sampling + segment conditioners ---
  waypoints_seg<<<BATCH, 256, 0, stream>>>(ctxf, gsel, wpq, wtw, wtb, wow, wob,
                                           res_scale, es, spw, spb, env_scale,
                                           segw, segb, seghb);

  // --- decoder LSTM (dominant: ~1536 WMMAs/step/block x 360 steps) ---
  size_t dec_smem = (2 * 16 * 1024 + 2 * 16 * 256 + 32 + 2048 + 256) * sizeof(float) +
                    4 * 16 * 256 * sizeof(_Float16);
  decoder_lstm<<<BATCH / 16, 256, dec_smem, stream>>>(
      ctxh, seghb, w0pos, w0ctx, w0st, dwhh0, dwih1, dwhh1, dbih0, dbhh0,
      dbih1, dbhh1, ow, ob, pred);
}
